// WeightedLoraMoELinear_67508295958840
// MI455X (gfx1250) — matrix operand order repaired and checked by reading, not compiled, not run on hardware
//
#include <hip/hip_runtime.h>
#include <hip/hip_bf16.h>
#include <math.h>

typedef __bf16 bf16_t;
typedef bf16_t v16bf __attribute__((ext_vector_type(16)));
typedef bf16_t v8bf  __attribute__((ext_vector_type(8)));
typedef float  v8f   __attribute__((ext_vector_type(8)));

#define EPSF 1e-6f
#define MAXE 16

#define BM 128
#define BN 128
#define BK 64
#define LDSK 72   // padded row stride (elements): 144B = 36 banks -> conflict-free frags

static __device__ __forceinline__ bf16_t f2bf(float f) { return (bf16_t)f; }

static __device__ __forceinline__ v16bf combine16(v8bf lo, v8bf hi) {
  v16bf r;
#pragma unroll
  for (int i = 0; i < 8; ++i) { r[i] = lo[i]; r[i + 8] = hi[i]; }
  return r;
}

// A-operand: 16x32 bf16 tile (row-major, leading dim ld)
// lanes 0-15: M=lane, K=0..7 then 16..23 ; lanes 16-31: M=lane-16, K=8..15 then 24..31
static __device__ __forceinline__ v16bf load_a_frag(const bf16_t* tile, int ld, int lane) {
  int rr = lane & 15, h = lane >> 4;
  const bf16_t* p = tile + (size_t)rr * ld + h * 8;
  v8bf lo = *(const v8bf*)(p);
  v8bf hi = *(const v8bf*)(p + 16);
  return combine16(lo, hi);
}

// B-operand from row-major [N][K]: lanes 0-15: N=lane, K=0..7 then 16..23 ; lanes 16-31: K=8..15 then 24..31
static __device__ __forceinline__ v16bf load_b_frag(const bf16_t* tile, int ld, int lane) {
  int rr = lane & 15, h = lane >> 4;
  const bf16_t* p = tile + (size_t)rr * ld + h * 8;
  v8bf lo = *(const v8bf*)(p);
  v8bf hi = *(const v8bf*)(p + 16);
  return combine16(lo, hi);
}

// async copy 16B from global (sbase + voff bytes) into LDS byte offset lds_off
static __device__ __forceinline__ void async_load_b128(unsigned lds_off, unsigned voff,
                                                       const bf16_t* sbase) {
  asm volatile("global_load_async_to_lds_b128 %0, %1, %2"
               :
               : "v"(lds_off), "v"(voff), "s"(sbase)
               : "memory");
}

static __device__ __forceinline__ void wait_async0() {
  asm volatile("s_wait_asynccnt 0" ::: "memory");
}

// ---------------- elementwise / conversion kernels ----------------

__global__ void k_f32_to_bf16(const float* __restrict__ src, bf16_t* __restrict__ dst, int n) {
  for (int i = blockIdx.x * blockDim.x + threadIdx.x; i < n; i += gridDim.x * blockDim.x)
    dst[i] = f2bf(src[i]);
}

// B [E][d_out][r] -> Bcat [d_out][E*r] (bf16)
__global__ void k_make_bcat(const float* __restrict__ B, bf16_t* __restrict__ Bcat,
                            int E, int dOut, int r) {
  int n = E * dOut * r;
  for (int i = blockIdx.x * blockDim.x + threadIdx.x; i < n; i += gridDim.x * blockDim.x) {
    int rr = i % r;
    int o  = (i / r) % dOut;
    int e  = i / (r * dOut);
    Bcat[(size_t)o * (E * r) + e * r + rr] = f2bf(B[i]);
  }
}

// rvn[e] = 1/(||gate_v[e]|| + eps)
__global__ void k_gate_norm(const float* __restrict__ gate_v, float* __restrict__ rvn, int d) {
  int e = blockIdx.x;
  const float* v = gate_v + (size_t)e * d;
  float s = 0.f;
  for (int i = threadIdx.x; i < d; i += blockDim.x) { float t = v[i]; s += t * t; }
  for (int off = 16; off; off >>= 1) s += __shfl_down(s, off);
  __shared__ float wsum[8];
  int wid = threadIdx.x >> 5, lane = threadIdx.x & 31;
  if (lane == 0) wsum[wid] = s;
  __syncthreads();
  if (threadIdx.x == 0) {
    float tot = 0.f; int nw = blockDim.x >> 5;
    for (int w = 0; w < nw; ++w) tot += wsum[w];
    rvn[e] = 1.0f / (sqrtf(tot) + EPSF);
  }
}

// Vnb[e][i] = bf16(gate_v[e][i] * rvn[e] / sqrt(d))
__global__ void k_make_vnb(const float* __restrict__ gate_v, const float* __restrict__ rvn,
                           bf16_t* __restrict__ Vnb, int E, int d) {
  int n = E * d;
  float rsd = rsqrtf((float)d);
  for (int i = blockIdx.x * blockDim.x + threadIdx.x; i < n; i += gridDim.x * blockDim.x)
    Vnb[i] = f2bf(gate_v[i] * rvn[i / d] * rsd);
}

// xinv[t] = 1/(||x_t|| + eps)
__global__ void k_xnorm(const float* __restrict__ x, float* __restrict__ xinv, int d) {
  int t = blockIdx.x;
  const float* xt = x + (size_t)t * d;
  float s = 0.f;
  for (int i = threadIdx.x; i < d; i += blockDim.x) { float v = xt[i]; s += v * v; }
  for (int off = 16; off; off >>= 1) s += __shfl_down(s, off);
  __shared__ float wsum[8];
  int wid = threadIdx.x >> 5, lane = threadIdx.x & 31;
  if (lane == 0) wsum[wid] = s;
  __syncthreads();
  if (threadIdx.x == 0) {
    float tot = 0.f; int nw = blockDim.x >> 5;
    for (int w = 0; w < nw; ++w) tot += wsum[w];
    xinv[t] = 1.0f / (sqrtf(tot) + EPSF);
  }
}

// ---------------- WMMA kernels ----------------

// scores_raw[T][E] = Xb @ Vnb^T   (block = 128 rows of T, 8 waves x one 16x16 tile)
__global__ void k_scores(const bf16_t* __restrict__ Xb, const bf16_t* __restrict__ Vnb,
                         float* __restrict__ scores, int K, int E) {
  int lane = threadIdx.x & 31, wid = threadIdx.x >> 5;
  int m0 = blockIdx.x * 128 + wid * 16;
  v8f acc;
#pragma unroll
  for (int i = 0; i < 8; ++i) acc[i] = 0.f;
  for (int k = 0; k < K; k += 32) {
    v16bf a = load_a_frag(Xb + (size_t)m0 * K + k, K, lane);
    v16bf b = load_b_frag(Vnb + k, K, lane);
    acc = __builtin_amdgcn_wmma_f32_16x16x32_bf16(false, a, false, b, (short)0, acc, false, false);
  }
  int col = lane & 15, h = lane >> 4;
  if (col < E) {
#pragma unroll
    for (int i = 0; i < 8; ++i) {
      int row = m0 + h * 8 + i;
      scores[(size_t)row * E + col] = acc[i];
    }
  }
}

// per-token top-k + softmax -> dense wte[t][e] = w * alpha[e]/r
__global__ void k_route(const float* __restrict__ scores, const float* __restrict__ xinv,
                        const float* __restrict__ alphas, const int* __restrict__ topk_p,
                        float* __restrict__ wte, int T, int E, int r) {
  int t = blockIdx.x * blockDim.x + threadIdx.x;
  if (t >= T) return;
  float sc[MAXE];
  float xi = xinv[t];
  for (int e = 0; e < E; ++e) sc[e] = scores[(size_t)t * E + e] * xi;
  int k = topk_p[0];
  if (k > E) k = E;
  float w[MAXE]; int idx[MAXE]; bool sel[MAXE];
  for (int e = 0; e < E; ++e) sel[e] = false;
  for (int j = 0; j < k; ++j) {
    int bi = 0; float bv = -3.4e38f;
    for (int e = 0; e < E; ++e)
      if (!sel[e] && sc[e] > bv) { bv = sc[e]; bi = e; }
    sel[bi] = true; idx[j] = bi; w[j] = bv;
  }
  float m = w[0], Z = 0.f;
  for (int j = 0; j < k; ++j) { w[j] = __expf(w[j] - m); Z += w[j]; }
  for (int e = 0; e < E; ++e) wte[(size_t)t * E + e] = 0.f;
  for (int j = 0; j < k; ++j)
    wte[(size_t)t * E + idx[j]] = (w[j] / Z) * (alphas[idx[j]] / (float)r);
}

// wmid[T][ER] = bf16( (Xb @ Acat^T)[t,c] * wte[t, c/r] )  (direct-global: small GEMM)
__global__ void k_mid(const bf16_t* __restrict__ Xb, const bf16_t* __restrict__ Acat,
                      const float* __restrict__ wte, bf16_t* __restrict__ wmid,
                      int K, int ER, int E, int r) {
  int lane = threadIdx.x & 31, wid = threadIdx.x >> 5;
  int m0 = blockIdx.y * 128 + (wid & 1) * 64;
  int n0 = blockIdx.x * 128 + (wid >> 1) * 32;
  v8f acc[4][2];
#pragma unroll
  for (int mi = 0; mi < 4; ++mi)
#pragma unroll
    for (int ni = 0; ni < 2; ++ni)
#pragma unroll
      for (int i = 0; i < 8; ++i) acc[mi][ni][i] = 0.f;

  for (int k = 0; k < K; k += 32) {
    v16bf a[4], b[2];
#pragma unroll
    for (int mi = 0; mi < 4; ++mi)
      a[mi] = load_a_frag(Xb + (size_t)(m0 + mi * 16) * K + k, K, lane);
#pragma unroll
    for (int ni = 0; ni < 2; ++ni)
      b[ni] = load_b_frag(Acat + (size_t)(n0 + ni * 16) * K + k, K, lane);
#pragma unroll
    for (int mi = 0; mi < 4; ++mi)
#pragma unroll
      for (int ni = 0; ni < 2; ++ni)
        acc[mi][ni] = __builtin_amdgcn_wmma_f32_16x16x32_bf16(
            false, a[mi], false, b[ni], (short)0, acc[mi][ni], false, false);
  }
  int nl = lane & 15, h = lane >> 4;
#pragma unroll
  for (int mi = 0; mi < 4; ++mi)
#pragma unroll
    for (int ni = 0; ni < 2; ++ni) {
      int col = n0 + ni * 16 + nl;
      int e = col / r;
#pragma unroll
      for (int i = 0; i < 8; ++i) {
        int row = m0 + mi * 16 + h * 8 + i;
        float v = acc[mi][ni][i] * wte[(size_t)row * E + e];
        wmid[(size_t)row * ER + col] = f2bf(v);
      }
    }
}

// out[T][N] = Xb @ Wb^T + bias + wmid @ Bcat^T
// async-to-LDS double-buffered staging; LoRA K-loop fused as extra staged iterations.
__global__ void k_main(const bf16_t* __restrict__ Xb, const bf16_t* __restrict__ Wb,
                       const float* __restrict__ bias,
                       const bf16_t* __restrict__ wmid, const bf16_t* __restrict__ Bcat,
                       float* __restrict__ out, int K, int N, int ER) {
  __shared__ bf16_t smem[2][2][BM][LDSK];  // [buf][A/B][row][col]

  int tid = threadIdx.x;
  int lane = tid & 31, wid = tid >> 5;
  int m0 = blockIdx.y * BM;
  int n0 = blockIdx.x * BN;
  int wm = (wid & 1) * 64;
  int wn = (wid >> 1) * 32;

  v8f acc[4][2];
#pragma unroll
  for (int mi = 0; mi < 4; ++mi)
#pragma unroll
    for (int ni = 0; ni < 2; ++ni)
#pragma unroll
      for (int i = 0; i < 8; ++i) acc[mi][ni][i] = 0.f;

  const bf16_t* baseA1 = Xb + (size_t)m0 * K;
  const bf16_t* baseB1 = Wb + (size_t)n0 * K;
  const bf16_t* baseA2 = wmid + (size_t)m0 * ER;
  const bf16_t* baseB2 = Bcat + (size_t)n0 * ER;

  int rowc  = tid >> 3;   // 0..31 (each thread covers 4 rows, stride 32)
  int chunk = tid & 7;    // 8 bf16 (16B) per chunk, 8 chunks per 64-wide row

  unsigned smem_base = (unsigned)(uintptr_t)(&smem[0][0][0][0]);

  auto issue = [&](int buf, const bf16_t* bA, const bf16_t* bB, int ld, int kbase) {
#pragma unroll
    for (int rr = 0; rr < 4; ++rr) {
      int row = rowc + rr * 32;
      unsigned va = (unsigned)(((unsigned)row * (unsigned)ld +
                                (unsigned)kbase + (unsigned)chunk * 8u) * 2u);
      unsigned la = smem_base +
          (unsigned)(((buf * 2 + 0) * BM + row) * LDSK + chunk * 8) * 2u;
      async_load_b128(la, va, bA);
      unsigned lb = smem_base +
          (unsigned)(((buf * 2 + 1) * BM + row) * LDSK + chunk * 8) * 2u;
      async_load_b128(lb, va, bB);
    }
  };

  const int nk1 = K / BK;
  const int nk2 = ER / BK;
  const int nk  = nk1 + nk2;

  issue(0, baseA1, baseB1, K, 0);

  for (int i = 0; i < nk; ++i) {
    int buf = i & 1;
    wait_async0();       // our async writes into smem[buf] are done
    __syncthreads();     // everyone's are done

    int j = i + 1;
    if (j < nk) {
      if (j < nk1) issue(j & 1, baseA1, baseB1, K, j * BK);
      else         issue(j & 1, baseA2, baseB2, ER, (j - nk1) * BK);
    }

#pragma unroll
    for (int kk = 0; kk < BK; kk += 32) {
      v16bf a[4], b[2];
#pragma unroll
      for (int mi = 0; mi < 4; ++mi)
        a[mi] = load_a_frag(&smem[buf][0][wm + mi * 16][kk], LDSK, lane);
#pragma unroll
      for (int ni = 0; ni < 2; ++ni)
        b[ni] = load_b_frag(&smem[buf][1][wn + ni * 16][kk], LDSK, lane);
#pragma unroll
      for (int mi = 0; mi < 4; ++mi)
#pragma unroll
        for (int ni = 0; ni < 2; ++ni)
          acc[mi][ni] = __builtin_amdgcn_wmma_f32_16x16x32_bf16(
              false, a[mi], false, b[ni], (short)0, acc[mi][ni], false, false);
    }
    __syncthreads();     // all reads of smem[buf] done before it is refilled
  }

  int nl = lane & 15, h = lane >> 4;
#pragma unroll
  for (int mi = 0; mi < 4; ++mi)
#pragma unroll
    for (int ni = 0; ni < 2; ++ni) {
      int col = n0 + wn + ni * 16 + nl;
      float bv = bias[col];
#pragma unroll
      for (int i = 0; i < 8; ++i) {
        int row = m0 + wm + mi * 16 + h * 8 + i;
        out[(size_t)row * N + col] = acc[mi][ni][i] + bv;
      }
    }
}

// ---------------- host launch ----------------

extern "C" void kernel_launch(void* const* d_in, const int* in_sizes, int n_in,
                              void* d_out, int out_size, void* d_ws, size_t ws_size,
                              hipStream_t stream) {
  const float* x      = (const float*)d_in[0];
  const float* base_W = (const float*)d_in[1];
  const float* base_b = (const float*)d_in[2];
  const float* A      = (const float*)d_in[3];
  const float* B      = (const float*)d_in[4];
  const float* gate_v = (const float*)d_in[5];
  const float* alphas = (const float*)d_in[6];
  const int*   topk   = (const int*)d_in[7];
  float* out = (float*)d_out;

  const int E    = in_sizes[6];
  const int d    = in_sizes[5] / E;
  const int r    = in_sizes[3] / (E * d);
  const int dOut = in_sizes[1] / d;
  const int T    = in_sizes[0] / d;
  const int ER   = E * r;

  char* p = (char*)d_ws;
  auto alloc = [&](size_t bytes) -> char* {
    char* q = p; p += (bytes + 255) & ~(size_t)255; return q;
  };
  bf16_t* Xb     = (bf16_t*)alloc((size_t)T * d * 2);
  bf16_t* Wb     = (bf16_t*)alloc((size_t)dOut * d * 2);
  bf16_t* Acat   = (bf16_t*)alloc((size_t)ER * d * 2);
  bf16_t* Bcat   = (bf16_t*)alloc((size_t)dOut * ER * 2);
  bf16_t* Vnb    = (bf16_t*)alloc((size_t)E * d * 2);
  bf16_t* wmid   = (bf16_t*)alloc((size_t)T * ER * 2);
  float*  scores = (float*)alloc((size_t)T * E * 4);
  float*  wte    = (float*)alloc((size_t)T * E * 4);
  float*  xinv   = (float*)alloc((size_t)T * 4);
  float*  rvn    = (float*)alloc((size_t)E * 4);

  const int TPB = 256;
  auto gs = [&](long long n) -> int {
    long long b = (n + TPB - 1) / TPB;
    return (int)(b > 4096 ? 4096 : b);
  };

  // conversions / small precompute
  k_gate_norm<<<E, TPB, 0, stream>>>(gate_v, rvn, d);
  k_make_vnb<<<gs((long long)E * d), TPB, 0, stream>>>(gate_v, rvn, Vnb, E, d);
  k_f32_to_bf16<<<gs((long long)T * d), TPB, 0, stream>>>(x, Xb, T * d);
  k_f32_to_bf16<<<gs((long long)dOut * d), TPB, 0, stream>>>(base_W, Wb, dOut * d);
  k_f32_to_bf16<<<gs((long long)ER * d), TPB, 0, stream>>>(A, Acat, ER * d);
  k_make_bcat<<<gs((long long)E * dOut * r), TPB, 0, stream>>>(B, Bcat, E, dOut, r);
  k_xnorm<<<T, TPB, 0, stream>>>(x, xinv, d);

  // routing
  k_scores<<<T / 128, TPB, 0, stream>>>(Xb, Vnb, scores, d, E);
  k_route<<<(T + TPB - 1) / TPB, TPB, 0, stream>>>(scores, xinv, alphas, topk, wte, T, E, r);

  // LoRA mid (weighted), then fused main GEMM
  dim3 gmid(ER / 128, T / 128);
  k_mid<<<gmid, TPB, 0, stream>>>(Xb, Acat, wte, wmid, d, ER, E, r);
  dim3 gmain(dOut / 128, T / 128);
  k_main<<<gmain, TPB, 0, stream>>>(Xb, Wb, base_b, wmid, Bcat, out, d, dOut, ER);
}
